// MultiheadSelfAttention_42468636623182
// MI455X (gfx1250) — matrix operand order repaired and checked by reading, not compiled, not run on hardware
//
#include <hip/hip_runtime.h>

// ---------------------------------------------------------------------------
// MI455X (gfx1250) MHSA forward, bf16 WMMA everywhere, f32 accumulate.
// convert x -> bf16 ; W -> bf16 transposed ; 3x GEMM (double-buffered LDS,
// async-to-LDS) scattering to [B,H,S,DK] ; RoPE ; causal flash attention
// (v_wmma_f32_16x16x32_bf16) ; final GEMM -> f32 d_out.
// ---------------------------------------------------------------------------

typedef unsigned short u16;
typedef unsigned int   u32;
typedef __attribute__((ext_vector_type(16))) __bf16 v16bf;
typedef __attribute__((ext_vector_type(8)))  float  v8f;
typedef int g4i __attribute__((vector_size(16)));   // matches builtin param type

union FragBF { uint4 u[2]; v16bf v; };   // 32 bytes = 8 VGPRs (A/B fragment)
union U4x8   { uint4 u;    u16 s[8]; };

#define B_    4
#define S_    2048
#define H_    16
#define DK_   128
#define D_    2048
#define M_    (B_*S_)          // 8192

#if __has_builtin(__builtin_amdgcn_global_load_async_to_lds_b128)
#define HAS_ASYNC 1
#else
#define HAS_ASYNC 0
#endif

__device__ __forceinline__ float bf2f(u32 u) {
  u <<= 16;
  return __builtin_bit_cast(float, u);
}
__device__ __forceinline__ u16 f2bf(float f) {        // round-to-nearest-even
  u32 u = __builtin_bit_cast(u32, f);
  u32 r = u + 0x7fffu + ((u >> 16) & 1u);
  return (u16)(r >> 16);
}
__device__ __forceinline__ v8f vzero8() {
  v8f z;
#pragma unroll
  for (int i = 0; i < 8; ++i) z[i] = 0.0f;
  return z;
}

// 16B global -> LDS copy: async (GLOBAL_LOAD_ASYNC_TO_LDS_B128) when available.
__device__ __forceinline__ void cp_b128(u16* l, const u16* g) {
#if HAS_ASYNC
  __builtin_amdgcn_global_load_async_to_lds_b128(
      (__attribute__((address_space(1))) g4i*)(u16*)g,
      (__attribute__((address_space(3))) g4i*)l, 0, 0);
#else
  *(uint4*)l = *(const uint4*)g;
#endif
}
__device__ __forceinline__ void cp_wait() {
#if HAS_ASYNC
#if __has_builtin(__builtin_amdgcn_s_wait_asynccnt)
  __builtin_amdgcn_s_wait_asynccnt(0);
#else
  asm volatile("s_wait_asynccnt 0x0" ::: "memory");
#endif
#endif
}

// ---------------------------------------------------------------------------
// Kernel 1: fp32 -> bf16, 8 elements / thread, b128 in / b128 out.
// ---------------------------------------------------------------------------
__global__ __launch_bounds__(256) void convert_f32_bf16(
    const float* __restrict__ src, u16* __restrict__ dst, int n) {
  int i = (blockIdx.x * 256 + threadIdx.x) * 8;
  if (i + 7 < n) {
    float4 a = *(const float4*)(src + i);
    float4 b = *(const float4*)(src + i + 4);
    uint4 o;
    o.x = (u32)f2bf(a.x) | ((u32)f2bf(a.y) << 16);
    o.y = (u32)f2bf(a.z) | ((u32)f2bf(a.w) << 16);
    o.z = (u32)f2bf(b.x) | ((u32)f2bf(b.y) << 16);
    o.w = (u32)f2bf(b.z) | ((u32)f2bf(b.w) << 16);
    *(uint4*)(dst + i) = o;
  }
}

// ---------------------------------------------------------------------------
// Kernel 2: W[k][n] fp32 (2048x2048) -> Wt[n][k] bf16 (transpose + convert).
// ---------------------------------------------------------------------------
__global__ __launch_bounds__(256) void transpose_convert(
    const float* __restrict__ src, u16* __restrict__ dst) {
  __shared__ float tile[32][33];
  int x = blockIdx.x * 32 + threadIdx.x;           // n
#pragma unroll
  for (int j = 0; j < 4; ++j) {
    int y = blockIdx.y * 32 + threadIdx.y + j * 8; // k
    tile[threadIdx.y + j * 8][threadIdx.x] = src[(size_t)y * D_ + x];
  }
  __syncthreads();
  int xo = blockIdx.y * 32 + threadIdx.x;          // k
#pragma unroll
  for (int j = 0; j < 4; ++j) {
    int yo = blockIdx.x * 32 + threadIdx.y + j * 8; // n
    dst[(size_t)yo * D_ + xo] = f2bf(tile[threadIdx.x][threadIdx.y + j * 8]);
  }
}

// ---------------------------------------------------------------------------
// Kernel 3: bf16 GEMM  C[M,N] = A[M,K] * Bt[N,K]^T,  K = N = 2048.
// Block tile 128x256, BK=32, double-buffered LDS fed by async-to-LDS.
// 8 waves in a 2x4 grid, wave tile 64x64 -> 16 wmma / 16 b128 frag loads.
// MODE 0: bf16 epilogue scattered to [B,H,S,DK] (QKV).
// MODE 1: f32 epilogue, plain row-major [M,N] (final projection).
// ---------------------------------------------------------------------------
template <int MODE>
__global__ __launch_bounds__(256) void gemm_bf16(
    const u16* __restrict__ A, const u16* __restrict__ Bt,
    void* __restrict__ out) {
  __shared__ __align__(16) u16 As[2][128][40];   // 32 K + 8 pad
  __shared__ __align__(16) u16 Bs[2][256][40];

  const int t = threadIdx.x, w = t >> 5, l = t & 31;
  const int half = l >> 4, ln = l & 15;
  const int m0 = blockIdx.y * 128, n0 = blockIdx.x * 256;
  const int wm = w >> 2, wn = w & 3;             // 2x4 wave grid, 64x64 tiles

  v8f acc[4][4];
#pragma unroll
  for (int mi = 0; mi < 4; ++mi)
#pragma unroll
    for (int ni = 0; ni < 4; ++ni) acc[mi][ni] = vzero8();

  // --- preload stage 0 ---
#pragma unroll
  for (int j = 0; j < 2; ++j) {                  // A: 512 x 16B
    int idx = t + j * 256, row = idx >> 2, c = idx & 3;
    cp_b128(&As[0][row][c * 8], &A[(size_t)(m0 + row) * D_ + c * 8]);
  }
#pragma unroll
  for (int j = 0; j < 4; ++j) {                  // B: 1024 x 16B
    int idx = t + j * 256, row = idx >> 2, c = idx & 3;
    cp_b128(&Bs[0][row][c * 8], &Bt[(size_t)(n0 + row) * D_ + c * 8]);
  }
  cp_wait();
  __syncthreads();

  for (int kk = 0; kk < D_; kk += 32) {
    const int st = (kk >> 5) & 1;
    const int nx = st ^ 1;
    const bool hasNext = (kk + 32) < D_;

    if (hasNext) {                               // stream next stage (async)
      const int kn = kk + 32;
#pragma unroll
      for (int j = 0; j < 2; ++j) {
        int idx = t + j * 256, row = idx >> 2, c = idx & 3;
        cp_b128(&As[nx][row][c * 8], &A[(size_t)(m0 + row) * D_ + kn + c * 8]);
      }
#pragma unroll
      for (int j = 0; j < 4; ++j) {
        int idx = t + j * 256, row = idx >> 2, c = idx & 3;
        cp_b128(&Bs[nx][row][c * 8], &Bt[(size_t)(n0 + row) * D_ + kn + c * 8]);
      }
    }

    FragBF a[4], b[4];
#pragma unroll
    for (int mi = 0; mi < 4; ++mi) {             // A frag: row=ln, K split by lane-half
      int r = wm * 64 + mi * 16 + ln;
      a[mi].u[0] = *(const uint4*)&As[st][r][half * 8];
      a[mi].u[1] = *(const uint4*)&As[st][r][16 + half * 8];
    }
#pragma unroll
    for (int ni = 0; ni < 4; ++ni) {             // B frag: col=ln, K split by lane-half
      int c = wn * 64 + ni * 16 + ln;
      b[ni].u[0] = *(const uint4*)&Bs[st][c][half * 8];
      b[ni].u[1] = *(const uint4*)&Bs[st][c][16 + half * 8];
    }
#pragma unroll
    for (int mi = 0; mi < 4; ++mi)
#pragma unroll
      for (int ni = 0; ni < 4; ++ni)
        acc[mi][ni] = __builtin_amdgcn_wmma_f32_16x16x32_bf16(
            false, a[mi].v, false, b[ni].v, (short)0, acc[mi][ni], false, false);

    cp_wait();
    __syncthreads();
  }

  // epilogue: C layout -> VGPR i holds row i+8*half, col = ln
#pragma unroll
  for (int mi = 0; mi < 4; ++mi)
#pragma unroll
    for (int ni = 0; ni < 4; ++ni)
#pragma unroll
      for (int i = 0; i < 8; ++i) {
        int row = m0 + wm * 64 + mi * 16 + i + 8 * half;
        int col = n0 + wn * 64 + ni * 16 + ln;
        float v = acc[mi][ni][i];
        if (MODE == 0) {
          int b_ = row >> 11, s = row & (S_ - 1);
          int h  = col >> 7,  dk = col & (DK_ - 1);
          ((u16*)out)[(((size_t)(b_ * H_ + h)) * S_ + s) * DK_ + dk] = f2bf(v);
        } else {
          ((float*)out)[(size_t)row * D_ + col] = v;
        }
      }
}

// ---------------------------------------------------------------------------
// Kernel 4: RoPE in-place on bf16 [B,H,S,DK]; one thread per (row, pair).
// ---------------------------------------------------------------------------
__global__ __launch_bounds__(256) void rope_kernel(
    u16* __restrict__ x, const int* __restrict__ pos) {
  int idx = blockIdx.x * 256 + threadIdx.x;       // < B*H*S*64
  if (idx >= B_ * H_ * S_ * 64) return;
  int j   = idx & 63;                              // pair index
  int row = idx >> 6;                              // (b*H+h)*S + s
  int s   = row & (S_ - 1);
  float p = (float)pos[s];
  float ang = p * expf((float)j * -0.14391156608f); // theta^(-2j/dk)
  float sn, cs;
  sincosf(ang, &sn, &cs);
  u32* ptr = (u32*)(x + (size_t)row * DK_ + 2 * j);
  u32 pv = *ptr;
  float x0 = bf2f(pv & 0xffffu), x1 = bf2f(pv >> 16);
  *ptr = (u32)f2bf(x0 * cs - x1 * sn) | ((u32)f2bf(x1 * cs + x0 * sn) << 16);
}

// ---------------------------------------------------------------------------
// Kernel 5: causal flash attention, bf16 WMMA, f32 accum.
// Grid: (S/128, B*H). Block 256 = 8 waves; wave owns 16 query rows.
// ---------------------------------------------------------------------------
__global__ __launch_bounds__(256) void flash_attn(
    const u16* __restrict__ Q, const u16* __restrict__ Kp,
    const u16* __restrict__ Vp, u16* __restrict__ Oattn) {
  __shared__ __align__(16) u16 Qs[128][136];       // 128 dk + 8 pad
  __shared__ __align__(16) u16 Ks[32][136];
  __shared__ __align__(16) u16 Vt[128][40];        // [dk][kv], 32 kv + 8 pad
  __shared__ __align__(16) u16 Ps[8][16][40];      // per-wave P scratch

  const int t = threadIdx.x, w = t >> 5, l = t & 31;
  const int half = l >> 4, ln = l & 15;
  const int qb = blockIdx.x * 128;
  const int bh = blockIdx.y;
  const u16* Qb = Q  + (size_t)bh * S_ * DK_;
  const u16* Kb = Kp + (size_t)bh * S_ * DK_;
  const u16* Vb = Vp + (size_t)bh * S_ * DK_;

  // stage the 128x128 Q block once (async; first cp_wait covers it)
#pragma unroll
  for (int j = 0; j < 8; ++j) {
    int idx = t + j * 256;                         // 2048 x uint4
    int row = idx >> 4, c = idx & 15;
    cp_b128(&Qs[row][c * 8], &Qb[(size_t)(qb + row) * DK_ + c * 8]);
  }

  float m_i[8], l_i[8];
  v8f acc[8];
#pragma unroll
  for (int i = 0; i < 8; ++i) { m_i[i] = -1e30f; l_i[i] = 0.0f; acc[i] = vzero8(); }

  const float scale = 0.08838834764831845f;        // 1/sqrt(128)
  const int sq_base = qb + w * 16;
  const int nkv = (qb >> 5) + 4;                   // causal extent in 32-key blocks

  for (int kv = 0; kv < nkv; ++kv) {
    const int kv0 = kv * 32;
    __syncthreads();                               // prior Ks/Vt/Ps reads done
    // K tile 32x128 (async)
#pragma unroll
    for (int j = 0; j < 2; ++j) {
      int idx = t + j * 256;
      int row = idx >> 4, c = idx & 15;
      cp_b128(&Ks[row][c * 8], &Kb[(size_t)(kv0 + row) * DK_ + c * 8]);
    }
    // V tile transposed into [dk][kv] (needs element shuffle -> manual)
    {
      int k = t >> 3, seg = t & 7, dk0 = seg * 16;
      U4x8 v0, v1;
      v0.u = *(const uint4*)&Vb[(size_t)(kv0 + k) * DK_ + dk0];
      v1.u = *(const uint4*)&Vb[(size_t)(kv0 + k) * DK_ + dk0 + 8];
#pragma unroll
      for (int e = 0; e < 8; ++e) Vt[dk0 + e][k]     = v0.s[e];
#pragma unroll
      for (int e = 0; e < 8; ++e) Vt[dk0 + 8 + e][k] = v1.s[e];
    }
    // prefetch next K/V block into L2 while this one computes
    if (kv + 1 < nkv) {
      __builtin_prefetch(&Kb[(size_t)(kv0 + 32) * DK_ + t * 16], 0, 0);
      __builtin_prefetch(&Vb[(size_t)(kv0 + 32) * DK_ + t * 16], 0, 0);
    }
    cp_wait();
    __syncthreads();

    // ---- scores: S(16x32) = Qw(16x128) . Ktile(32x128)^T ----
    v8f sc[2];
    sc[0] = vzero8(); sc[1] = vzero8();
#pragma unroll
    for (int ks = 0; ks < 4; ++ks) {
      FragBF qa;
      int r = w * 16 + ln;
      qa.u[0] = *(const uint4*)&Qs[r][ks * 32 + half * 8];
      qa.u[1] = *(const uint4*)&Qs[r][ks * 32 + 16 + half * 8];
#pragma unroll
      for (int tt = 0; tt < 2; ++tt) {
        FragBF kb;
        int n = tt * 16 + ln;
        kb.u[0] = *(const uint4*)&Ks[n][ks * 32 + half * 8];
        kb.u[1] = *(const uint4*)&Ks[n][ks * 32 + 16 + half * 8];
        sc[tt] = __builtin_amdgcn_wmma_f32_16x16x32_bf16(
            false, qa.v, false, kb.v, (short)0, sc[tt], false, false);
      }
    }

    // ---- online softmax (row lives across one 16-lane half) ----
    float alpha[8];
#pragma unroll
    for (int i = 0; i < 8; ++i) {
      int sq = sq_base + i + 8 * half;
      float x0 = sc[0][i] * scale; if (kv0 + ln      > sq) x0 = -1e30f;
      float x1 = sc[1][i] * scale; if (kv0 + 16 + ln > sq) x1 = -1e30f;
      float mx = fmaxf(x0, x1);
#pragma unroll
      for (int d = 1; d < 16; d <<= 1) mx = fmaxf(mx, __shfl_xor(mx, d, 32));
      float mnew = fmaxf(m_i[i], mx);
      alpha[i] = __expf(m_i[i] - mnew);
      m_i[i] = mnew;
      float p0 = __expf(x0 - mnew);
      float p1 = __expf(x1 - mnew);
      int r = i + 8 * half;
      Ps[w][r][ln]      = f2bf(p0);
      Ps[w][r][16 + ln] = f2bf(p1);
      float rs = p0 + p1;
#pragma unroll
      for (int d = 1; d < 16; d <<= 1) rs += __shfl_xor(rs, d, 32);
      l_i[i] = l_i[i] * alpha[i] + rs;
    }
#pragma unroll
    for (int nt = 0; nt < 8; ++nt)
#pragma unroll
      for (int i = 0; i < 8; ++i) acc[nt][i] *= alpha[i];

    __syncthreads();                               // P visible across lanes

    // ---- O += P(16x32) . Vtile(32x128) ----
    FragBF pa;
    pa.u[0] = *(const uint4*)&Ps[w][ln][half * 8];
    pa.u[1] = *(const uint4*)&Ps[w][ln][16 + half * 8];
#pragma unroll
    for (int nt = 0; nt < 8; ++nt) {
      FragBF vb;
      int n = nt * 16 + ln;
      vb.u[0] = *(const uint4*)&Vt[n][half * 8];
      vb.u[1] = *(const uint4*)&Vt[n][16 + half * 8];
      acc[nt] = __builtin_amdgcn_wmma_f32_16x16x32_bf16(
          false, pa.v, false, vb.v, (short)0, acc[nt], false, false);
    }
  }

  // ---- epilogue: O/l -> bf16 [B,S,H,DK] ----
  const int b_ = bh >> 4, h = bh & (H_ - 1);
#pragma unroll
  for (int i = 0; i < 8; ++i) {
    float inv = 1.0f / l_i[i];
    int s = qb + w * 16 + i + 8 * half;
#pragma unroll
    for (int nt = 0; nt < 8; ++nt) {
      int dk = nt * 16 + ln;
      Oattn[(((size_t)b_ * S_ + s) * H_ + h) * DK_ + dk] = f2bf(acc[nt][i] * inv);
    }
  }
}

// ---------------------------------------------------------------------------
// Host launcher
// ---------------------------------------------------------------------------
extern "C" void kernel_launch(void* const* d_in, const int* in_sizes, int n_in,
                              void* d_out, int out_size, void* d_ws, size_t ws_size,
                              hipStream_t stream) {
  (void)in_sizes; (void)n_in; (void)out_size; (void)ws_size;
  const float* x  = (const float*)d_in[0];
  const float* Wq = (const float*)d_in[1];
  const float* Wk = (const float*)d_in[2];
  const float* Wv = (const float*)d_in[3];
  const float* Wo = (const float*)d_in[4];
  const int*  pos = (const int*)d_in[5];

  char* ws = (char*)d_ws;
  const size_t SZ_X = (size_t)M_ * D_ * 2;             // 32 MiB
  const size_t SZ_W = (size_t)D_ * D_ * 2;             //  8 MiB
  const size_t SZ_H = (size_t)B_ * H_ * S_ * DK_ * 2;  // 32 MiB
  u16* xb   = (u16*)(ws);
  u16* wqT  = (u16*)(ws + SZ_X);
  u16* wkT  = (u16*)(ws + SZ_X + SZ_W);
  u16* wvT  = (u16*)(ws + SZ_X + 2 * SZ_W);
  u16* woT  = (u16*)(ws + SZ_X + 3 * SZ_W);
  u16* qh   = (u16*)(ws + SZ_X + 4 * SZ_W);
  u16* kh   = (u16*)(ws + SZ_X + 4 * SZ_W + SZ_H);
  u16* vh   = (u16*)(ws + SZ_X + 4 * SZ_W + 2 * SZ_H);
  u16* attn = (u16*)(ws + SZ_X + 4 * SZ_W + 3 * SZ_H);

  // 1) conversions
  convert_f32_bf16<<<(M_ * D_) / (256 * 8), 256, 0, stream>>>(x, xb, M_ * D_);
  dim3 tb(32, 8), tg(D_ / 32, D_ / 32);
  transpose_convert<<<tg, tb, 0, stream>>>(Wq, wqT);
  transpose_convert<<<tg, tb, 0, stream>>>(Wk, wkT);
  transpose_convert<<<tg, tb, 0, stream>>>(Wv, wvT);
  transpose_convert<<<tg, tb, 0, stream>>>(Wo, woT);

  // 2) QKV projections, scattered to head-major layout
  dim3 gg(D_ / 256, M_ / 128);
  gemm_bf16<0><<<gg, 256, 0, stream>>>(xb, wqT, qh);
  gemm_bf16<0><<<gg, 256, 0, stream>>>(xb, wkT, kh);
  gemm_bf16<0><<<gg, 256, 0, stream>>>(xb, wvT, vh);

  // 3) RoPE on Q and K
  rope_kernel<<<(B_ * H_ * S_ * 64) / 256, 256, 0, stream>>>(qh, pos);
  rope_kernel<<<(B_ * H_ * S_ * 64) / 256, 256, 0, stream>>>(kh, pos);

  // 4) causal flash attention
  flash_attn<<<dim3(S_ / 128, B_ * H_), 256, 0, stream>>>(qh, kh, vh, attn);

  // 5) output projection -> f32 d_out
  gemm_bf16<1><<<gg, 256, 0, stream>>>(attn, woT, (float*)d_out);
}
